// CrystalGraphEncoder_31714038514078
// MI455X (gfx1250) — matrix operand order repaired and checked by reading, not compiled, not run on hardware
//
#include <hip/hip_runtime.h>
#include <hip/hip_bf16.h>

// ---------------------------------------------------------------------------
// CrystalGraphEncoder for MI455X (gfx1250).
// GEMMs (the compute-bound ~28 GFLOP core) run on v_wmma_f32_16x16x32_bf16
// with fragment-major LDS layouts (ds_load_b128 operand fetch, float4 global
// staging), K templated so the k-loop is fully unrolled and branch-free.
// Workspace matrices are row-padded to 64 so epilogue stores are unguarded.
// Edge softmax/scatter is memory-bound -> VALU + float atomics.
// ---------------------------------------------------------------------------

#define IN_DIM 92
#define HID    256
#define HEADS  8
#define HDIM   32
#define NGRAPH 128
#define LN_EPS 1e-5f

#define FLAG_RELU 1
#define FLAG_ADDC 2

typedef __attribute__((ext_vector_type(16))) __bf16 v16bf;
typedef __attribute__((ext_vector_type(8)))  __bf16 v8bf;
typedef __attribute__((ext_vector_type(8)))  float  v8f;

// ---------------- utility kernels ----------------

__global__ void fill_f32(float* p, float v, int n) {
    int i = blockIdx.x * blockDim.x + threadIdx.x;
    if (i < n) p[i] = v;
}

__global__ void fill_u32(unsigned* p, unsigned v, int n) {
    int i = blockIdx.x * blockDim.x + threadIdx.x;
    if (i < n) p[i] = v;
}

__global__ void build_self_loops(const int* __restrict__ ei,
                                 int* __restrict__ src_sl, int* __restrict__ dst_sl,
                                 int E, int N) {
    int i = blockIdx.x * blockDim.x + threadIdx.x;
    int Esl = E + N;
    if (i >= Esl) return;
    if (i < E) { src_sl[i] = ei[i]; dst_sl[i] = ei[E + i]; }
    else       { int n = i - E; src_sl[i] = n; dst_sl[i] = n; }
}

// ---------------- WMMA bf16 GEMM: C[M,N] = A[M,K] @ W[K,N] (+bias)(+C)(relu) ----
// Block tile 64x64, 8 waves; wave (wm,wn) computes rows wm*16..+16, cols wn*32..+32
// as two 16x16 f32 accumulators fed by v_wmma_f32_16x16x32_bf16.
//
// LDS layouts are fragment-major so each lane's WMMA operand is one contiguous,
// 32B-aligned v16bf load:
//   As[row][k]: k in natural order; lane half h reads As[row][h*16 .. h*16+16).
//   Bs[col][k]: transposed tile; lane reads Bs[col][h*16 .. h*16+16).
//
// Row handling is branch-free: A-row reads clamp to Mrd-1 (dup rows only feed
// padded outputs), C rows are guaranteed in-range (grid.y == Mst/64 with Mst-
// padded buffers). K remainder (only K=92) is a peeled branchless step using
// clamped addresses + select-zero.

template <int K>
__global__ __launch_bounds__(256)
void gemm_bias_wmma(const float* __restrict__ A, const float* __restrict__ W,
                    const float* __restrict__ bias, float* __restrict__ C,
                    int Mrd, int Nout, int flags) {
    __shared__ __attribute__((aligned(32))) __bf16 As[64][32];
    __shared__ __attribute__((aligned(32))) __bf16 Bs[64][32];

    const int tid  = threadIdx.x;
    const int lane = tid & 31;
    const int wave = tid >> 5;
    const int wm   = wave >> 1;     // 0..3 : row group (16 rows)
    const int wn   = wave & 1;      // 0..1 : col group (32 cols)
    const int half = lane >> 4;     // 0/1
    const int l15  = lane & 15;

    const int rowBase = blockIdx.y * 64;
    const int colBase = blockIdx.x * 64;

    // A staging map: thread -> (row, 8-k block)
    const int ar   = tid >> 2;           // 0..63
    const int akc  = (tid & 3) * 8;      // 0,8,16,24
    const int apos = akc;

    int gr = rowBase + ar;
    if (gr >= Mrd) gr = Mrd - 1;         // clamped: dup row feeds only padded out
    const float* aRow = A + (size_t)gr * K;

    v8f acc0 = {}; v8f acc1 = {};

    constexpr int KFULL = K / 32;        // full 32-wide k-steps
    constexpr int KREM  = K % 32;        // remainder (only for K=92 -> 28)

    #pragma unroll
    for (int ks = 0; ks < KFULL + (KREM ? 1 : 0); ++ks) {
        const int k0 = ks * 32;
        const bool full = (ks < KFULL);

        // ---- stage A tile (64x32) fp32 -> bf16, natural-k layout ----
        {
            float f[8];
            if (full) {
                const float* ap = aRow + k0 + akc;
                float4 f0 = ((const float4*)ap)[0];
                float4 f1 = ((const float4*)ap)[1];
                f[0] = f0.x; f[1] = f0.y; f[2] = f0.z; f[3] = f0.w;
                f[4] = f1.x; f[5] = f1.y; f[6] = f1.z; f[7] = f1.w;
            } else {
                #pragma unroll
                for (int j = 0; j < 8; ++j) {
                    int gk = k0 + akc + j;
                    int gkc = gk < K ? gk : K - 1;          // always-valid address
                    float v = aRow[gkc];                    // unconditional load
                    f[j] = (gk < K) ? v : 0.f;              // cndmask, no branch
                }
            }
            v8bf av;
            #pragma unroll
            for (int j = 0; j < 8; ++j) av[j] = (__bf16)f[j];
            *(v8bf*)&As[ar][apos] = av;
        }

        // ---- stage B tile (32x64) fp32 -> bf16, transposed layout ----
        #pragma unroll
        for (int q = 0; q < 2; ++q) {
            const int idx4 = tid + q * 256;      // 0..511 float4 slots
            const int kk   = idx4 >> 4;          // 0..31
            const int cg   = (idx4 & 15) * 4;    // col group base
            const int gk   = k0 + kk;
            const int gkc  = gk < K ? gk : K - 1;
            const float* wp = W + (size_t)gkc * Nout + colBase + cg;
            float4 f0 = *(const float4*)wp;      // unconditional (row valid)
            float f[4];
            bool ok = full || (gk < K);
            f[0] = ok ? f0.x : 0.f; f[1] = ok ? f0.y : 0.f;
            f[2] = ok ? f0.z : 0.f; f[3] = ok ? f0.w : 0.f;
            #pragma unroll
            for (int j = 0; j < 4; ++j) Bs[cg + j][kk] = (__bf16)f[j];
        }
        __syncthreads();

        // ---- operand fetch: one contiguous 32B vector per fragment ----
        v16bf afrag  = *(const v16bf*)&As[wm * 16 + l15][half * 16];
        v16bf bfrag0 = *(const v16bf*)&Bs[wn * 32 + l15][half * 16];
        v16bf bfrag1 = *(const v16bf*)&Bs[wn * 32 + 16 + l15][half * 16];

        acc0 = __builtin_amdgcn_wmma_f32_16x16x32_bf16(false, afrag, false, bfrag0,
                                                       (short)0, acc0, false, false);
        acc1 = __builtin_amdgcn_wmma_f32_16x16x32_bf16(false, afrag, false, bfrag1,
                                                       (short)0, acc1, false, false);
        __syncthreads();
    }

    // ---- epilogue: C/D layout lane=N, VGPR v -> M = v + 8*half ----
    // Rows are guaranteed in-range (padded destination), cols exact: no guards.
    #pragma unroll
    for (int c = 0; c < 2; ++c) {
        #pragma unroll
        for (int v = 0; v < 8; ++v) {
            int row = rowBase + wm * 16 + v + half * 8;
            int col = colBase + wn * 32 + c * 16 + l15;
            float r = (c == 0) ? acc0[v] : acc1[v];
            if (bias)              r += bias[col];
            if (flags & FLAG_ADDC) r += C[(size_t)row * Nout + col];
            if (flags & FLAG_RELU) r = fmaxf(r, 0.f);
            C[(size_t)row * Nout + col] = r;
        }
    }
}

// ---------------- edge / segment-softmax kernels ----------------

__device__ inline unsigned f2key(float f) {
    unsigned u = __float_as_uint(f);
    return (u & 0x80000000u) ? ~u : (u | 0x80000000u);
}
__device__ inline float key2f(unsigned k) {
    unsigned u = (k & 0x80000000u) ? (k & 0x7FFFFFFFu) : ~k;
    return __uint_as_float(u);
}

__global__ void tc_logits(const float* __restrict__ q, const float* __restrict__ k,
                          const int* __restrict__ src, const int* __restrict__ dst,
                          float* __restrict__ logits, int E) {
    int idx = blockIdx.x * blockDim.x + threadIdx.x;
    if (idx >= E * HEADS) return;
    int e = idx >> 3, h = idx & 7;
    const float4* qp = (const float4*)(q + (size_t)dst[e] * HID + h * HDIM);
    const float4* kp = (const float4*)(k + (size_t)src[e] * HID + h * HDIM);
    float acc = 0.f;
    #pragma unroll
    for (int d = 0; d < HDIM / 4; ++d) {
        float4 a = qp[d], b = kp[d];
        acc += a.x * b.x + a.y * b.y + a.z * b.z + a.w * b.w;
    }
    logits[idx] = acc * 0.17677669529663687f;   // 1/sqrt(32)
}

__global__ void seg_max(const float* __restrict__ logits, const int* __restrict__ dst,
                        unsigned* __restrict__ maxkey, int E) {
    int idx = blockIdx.x * blockDim.x + threadIdx.x;
    if (idx >= E * HEADS) return;
    int e = idx >> 3, h = idx & 7;
    atomicMax(&maxkey[dst[e] * HEADS + h], f2key(logits[idx]));
}

__global__ void seg_expsum(const float* __restrict__ logits, const int* __restrict__ dst,
                           const unsigned* __restrict__ maxkey,
                           float* __restrict__ sums, float* __restrict__ wexp, int E) {
    int idx = blockIdx.x * blockDim.x + threadIdx.x;
    if (idx >= E * HEADS) return;
    int e = idx >> 3, h = idx & 7;
    float m = key2f(maxkey[dst[e] * HEADS + h]);
    float w = __expf(logits[idx] - m);
    wexp[idx] = w;
    atomicAdd(&sums[dst[e] * HEADS + h], w);
}

__global__ void seg_aggregate(const float* __restrict__ wexp, const float* __restrict__ sums,
                              const float* __restrict__ vals,
                              const int* __restrict__ src, const int* __restrict__ dst,
                              float* __restrict__ agg, int E) {
    int idx = blockIdx.x * blockDim.x + threadIdx.x;
    if (idx >= E * HEADS) return;
    int e = idx >> 3, h = idx & 7;
    float coef = wexp[idx] / (sums[dst[e] * HEADS + h] + 1e-16f);
    const float* vp = vals + (size_t)src[e] * HID + h * HDIM;
    float* ap = agg + (size_t)dst[e] * HID + h * HDIM;
    #pragma unroll
    for (int d = 0; d < HDIM; ++d) atomicAdd(&ap[d], coef * vp[d]);
}

__global__ void gat_scores(const float* __restrict__ hh,
                           const float* __restrict__ att_s, const float* __restrict__ att_d,
                           float* __restrict__ a_s, float* __restrict__ a_d, int N) {
    int idx = blockIdx.x * blockDim.x + threadIdx.x;
    if (idx >= N * HEADS) return;
    int n = idx >> 3, h = idx & 7;
    const float4* hp = (const float4*)(hh + (size_t)n * HID + h * HDIM);
    const float4* sp = (const float4*)(att_s + h * HDIM);
    const float4* dp = (const float4*)(att_d + h * HDIM);
    float s = 0.f, d = 0.f;
    #pragma unroll
    for (int k = 0; k < HDIM / 4; ++k) {
        float4 hv = hp[k], sv = sp[k], dv = dp[k];
        s += hv.x * sv.x + hv.y * sv.y + hv.z * sv.z + hv.w * sv.w;
        d += hv.x * dv.x + hv.y * dv.y + hv.z * dv.z + hv.w * dv.w;
    }
    a_s[idx] = s; a_d[idx] = d;
}

__global__ void gat_logits(const float* __restrict__ a_s, const float* __restrict__ a_d,
                           const int* __restrict__ src, const int* __restrict__ dst,
                           float* __restrict__ logits, int E) {
    int idx = blockIdx.x * blockDim.x + threadIdx.x;
    if (idx >= E * HEADS) return;
    int e = idx >> 3, h = idx & 7;
    float v = a_s[src[e] * HEADS + h] + a_d[dst[e] * HEADS + h];
    logits[idx] = (v > 0.f) ? v : 0.2f * v;     // leaky_relu(0.2)
}

// ---------------- residual + LayerNorm + ReLU (block per node) --------------

__global__ __launch_bounds__(HID)
void ln_res_relu(const float* __restrict__ outb, const float* __restrict__ extra_bias,
                 const float* __restrict__ g, const float* __restrict__ b,
                 float* __restrict__ h, int N) {
    int n = blockIdx.x;
    int t = threadIdx.x;
    __shared__ float red[HID];
    float val = outb[(size_t)n * HID + t] + h[(size_t)n * HID + t];
    if (extra_bias) val += extra_bias[t];
    red[t] = val; __syncthreads();
    for (int s = HID / 2; s > 0; s >>= 1) { if (t < s) red[t] += red[t + s]; __syncthreads(); }
    float mu = red[0] / (float)HID;
    __syncthreads();
    float d = val - mu;
    red[t] = d * d; __syncthreads();
    for (int s = HID / 2; s > 0; s >>= 1) { if (t < s) red[t] += red[t + s]; __syncthreads(); }
    float var = red[0] / (float)HID;
    float y = d * rsqrtf(var + LN_EPS) * g[t] + b[t];
    h[(size_t)n * HID + t] = fmaxf(y, 0.f);
}

// ---------------- pooling ----------------

__global__ void pool_sum(const float* __restrict__ h, const int* __restrict__ batch,
                         float* __restrict__ sums, float* __restrict__ cnt, int N) {
    int idx = blockIdx.x * blockDim.x + threadIdx.x;
    if (idx >= N * HID) return;
    int n = idx >> 8, f = idx & (HID - 1);
    int gidx = batch[n];
    atomicAdd(&sums[(size_t)gidx * HID + f], h[idx]);
    if (f == 0) atomicAdd(&cnt[gidx], 1.f);
}

__global__ void pool_div(float* __restrict__ sums, const float* __restrict__ cnt) {
    int idx = blockIdx.x * blockDim.x + threadIdx.x;
    if (idx >= NGRAPH * HID) return;
    int gidx = idx >> 8;
    sums[idx] = sums[idx] / fmaxf(cnt[gidx], 1.f);
}

// ---------------- host orchestration ----------------

static inline int cd(int a, int b) { return (a + b - 1) / b; }

extern "C" void kernel_launch(void* const* d_in, const int* in_sizes, int n_in,
                              void* d_out, int out_size, void* d_ws, size_t ws_size,
                              hipStream_t stream) {
    const int N    = in_sizes[0] / IN_DIM;
    const int Npad = cd(N, 64) * 64;          // row-padded node count for GEMM buffers
    const int E    = in_sizes[1] / 2;
    const int Esl  = E + N;

    const float* x      = (const float*)d_in[0];
    const int*   ei     = (const int*)d_in[1];
    const int*   batch  = (const int*)d_in[2];
    const float* Win    = (const float*)d_in[3];
    const float* b_in   = (const float*)d_in[4];
    const float* Wq     = (const float*)d_in[5];
    const float* Wk     = (const float*)d_in[6];
    const float* Wv     = (const float*)d_in[7];
    const float* Wskip  = (const float*)d_in[8];
    const float* bq     = (const float*)d_in[9];
    const float* bk     = (const float*)d_in[10];
    const float* bv     = (const float*)d_in[11];
    const float* bskip  = (const float*)d_in[12];
    const float* Wg     = (const float*)d_in[13];
    const float* att_src= (const float*)d_in[14];
    const float* att_dst= (const float*)d_in[15];
    const float* bg     = (const float*)d_in[16];
    const float* ln_g   = (const float*)d_in[17];
    const float* ln_b   = (const float*)d_in[18];
    const float* W1     = (const float*)d_in[19];
    const float* b1     = (const float*)d_in[20];
    const float* W2     = (const float*)d_in[21];
    const float* b2     = (const float*)d_in[22];

    // scratch carving (256B aligned); node matrices padded to Npad rows
    char* wp = (char*)d_ws;
    auto carve = [&](size_t bytes) -> void* {
        void* p = wp; wp += (bytes + 255) & ~(size_t)255; return p;
    };
    float*    hbuf   = (float*)carve((size_t)Npad * HID * 4);
    float*    outb   = (float*)carve((size_t)Npad * HID * 4);
    float*    qb     = (float*)carve((size_t)Npad * HID * 4);
    float*    kb     = (float*)carve((size_t)Npad * HID * 4);
    float*    vb     = (float*)carve((size_t)Npad * HID * 4);
    float*    logits = (float*)carve((size_t)Esl * HEADS * 4);
    float*    wexp   = (float*)carve((size_t)Esl * HEADS * 4);
    unsigned* maxkey = (unsigned*)carve((size_t)N * HEADS * 4);
    float*    sums   = (float*)carve((size_t)N * HEADS * 4);
    float*    a_s    = (float*)carve((size_t)N * HEADS * 4);
    float*    a_d    = (float*)carve((size_t)N * HEADS * 4);
    int*      src_sl = (int*)carve((size_t)Esl * 4);
    int*      dst_sl = (int*)carve((size_t)Esl * 4);
    float*    psum   = (float*)carve((size_t)NGRAPH * HID * 4);
    float*    cnt    = (float*)carve((size_t)NGRAPH * 4);
    float*    h1     = (float*)carve((size_t)NGRAPH * 2 * HID * 4);

    const dim3 blk(256);
    // Mrd: readable rows of A (clamp bound). Mst: stored rows of C (multiple of 64).
    auto gemm = [&](const float* A, const float* W, const float* bias, float* C,
                    int Mrd, int Mst, int K, int Nout, int flags) {
        dim3 grid(Nout / 64, Mst / 64);
        if (K == IN_DIM)
            gemm_bias_wmma<IN_DIM><<<grid, blk, 0, stream>>>(A, W, bias, C, Mrd, Nout, flags);
        else if (K == HID)
            gemm_bias_wmma<HID><<<grid, blk, 0, stream>>>(A, W, bias, C, Mrd, Nout, flags);
        else
            gemm_bias_wmma<2 * HID><<<grid, blk, 0, stream>>>(A, W, bias, C, Mrd, Nout, flags);
    };

    // input projection: h = x @ Win + b_in
    gemm(x, Win, b_in, hbuf, N, Npad, IN_DIM, HID, 0);

    build_self_loops<<<cd(Esl, 256), blk, 0, stream>>>(ei, src_sl, dst_sl, E, N);

    for (int layer = 0; layer < 4; ++layer) {
        if ((layer & 1) == 0) {
            // ---- TransformerConv ----
            const int t = layer / 2;
            const size_t wof = (size_t)t * HID * HID;
            const size_t bof = (size_t)t * HID;
            gemm(hbuf, Wq + wof, bq + bof, qb, Npad, Npad, HID, HID, 0);
            gemm(hbuf, Wk + wof, bk + bof, kb, Npad, Npad, HID, HID, 0);
            gemm(hbuf, Wv + wof, bv + bof, vb, Npad, Npad, HID, HID, 0);

            tc_logits<<<cd(E * HEADS, 256), blk, 0, stream>>>(qb, kb, ei, ei + E, logits, E);
            fill_u32<<<cd(N * HEADS, 256), blk, 0, stream>>>(maxkey, 0u, N * HEADS);
            seg_max<<<cd(E * HEADS, 256), blk, 0, stream>>>(logits, ei + E, maxkey, E);
            fill_f32<<<cd(N * HEADS, 256), blk, 0, stream>>>(sums, 0.f, N * HEADS);
            seg_expsum<<<cd(E * HEADS, 256), blk, 0, stream>>>(logits, ei + E, maxkey, sums, wexp, E);
            fill_f32<<<cd(N * HID, 256), blk, 0, stream>>>(outb, 0.f, N * HID);
            seg_aggregate<<<cd(E * HEADS, 256), blk, 0, stream>>>(wexp, sums, vb, ei, ei + E, outb, E);

            // outb += h @ Wskip + bskip
            gemm(hbuf, Wskip + wof, bskip + bof, outb, Npad, Npad, HID, HID, FLAG_ADDC);
            ln_res_relu<<<N, HID, 0, stream>>>(outb, nullptr,
                                               ln_g + layer * HID, ln_b + layer * HID, hbuf, N);
        } else {
            // ---- GATConv (with self-loops) ----
            const int g = layer / 2;
            const size_t wof = (size_t)g * HID * HID;
            gemm(hbuf, Wg + wof, nullptr, vb, Npad, Npad, HID, HID, 0);   // hh
            gat_scores<<<cd(N * HEADS, 256), blk, 0, stream>>>(vb,
                att_src + (size_t)g * HEADS * HDIM, att_dst + (size_t)g * HEADS * HDIM,
                a_s, a_d, N);
            gat_logits<<<cd(Esl * HEADS, 256), blk, 0, stream>>>(a_s, a_d, src_sl, dst_sl, logits, Esl);
            fill_u32<<<cd(N * HEADS, 256), blk, 0, stream>>>(maxkey, 0u, N * HEADS);
            seg_max<<<cd(Esl * HEADS, 256), blk, 0, stream>>>(logits, dst_sl, maxkey, Esl);
            fill_f32<<<cd(N * HEADS, 256), blk, 0, stream>>>(sums, 0.f, N * HEADS);
            seg_expsum<<<cd(Esl * HEADS, 256), blk, 0, stream>>>(logits, dst_sl, maxkey, sums, wexp, Esl);
            fill_f32<<<cd(N * HID, 256), blk, 0, stream>>>(outb, 0.f, N * HID);
            seg_aggregate<<<cd(Esl * HEADS, 256), blk, 0, stream>>>(wexp, sums, vb, src_sl, dst_sl, outb, Esl);

            ln_res_relu<<<N, HID, 0, stream>>>(outb, bg + (size_t)g * HID,
                                               ln_g + layer * HID, ln_b + layer * HID, hbuf, N);
        }
    }

    // global mean pool
    fill_f32<<<cd(NGRAPH * HID, 256), blk, 0, stream>>>(psum, 0.f, NGRAPH * HID);
    fill_f32<<<cd(NGRAPH, 256), blk, 0, stream>>>(cnt, 0.f, NGRAPH);
    pool_sum<<<cd(N * HID, 256), blk, 0, stream>>>(hbuf, batch, psum, cnt, N);
    pool_div<<<cd(NGRAPH * HID, 256), blk, 0, stream>>>(psum, cnt);

    // MLP head
    gemm(psum, W1, b1, h1, NGRAPH, NGRAPH, HID, 2 * HID, FLAG_RELU);
    gemm(h1, W2, b2, (float*)d_out, NGRAPH, NGRAPH, 2 * HID, 128, 0);
}
